// TransformerBlock_39384850104591
// MI455X (gfx1250) — matrix operand order repaired and checked by reading, not compiled, not run on hardware
//
#include <hip/hip_runtime.h>
#include <hip/hip_bf16.h>

// ---------------------------------------------------------------------------
// MLA transformer block for MI455X (gfx1250).
// All GEMMs run through v_wmma_f32_16x16x32_bf16 (fp32 accumulate).
// All LDS fragment traffic is 128-bit (pre-transposed B operands).
// Flash-attention K/V tiles staged via the Tensor Data Mover when available.
// ---------------------------------------------------------------------------

typedef __attribute__((ext_vector_type(16))) __bf16          v16bf;
typedef __attribute__((ext_vector_type(16))) unsigned short  v16u;
typedef __attribute__((ext_vector_type(8)))  unsigned short  v8u;
typedef __attribute__((ext_vector_type(8)))  float           v8f;

#if defined(__AMDGCN__) && __has_builtin(__builtin_amdgcn_tensor_load_to_lds) && __has_builtin(__builtin_amdgcn_s_wait_tensorcnt)
#define USE_TDM 1
#else
#define USE_TDM 0
#endif

static __device__ __forceinline__ unsigned short f32_to_bf16(float f) {
  unsigned int u = __float_as_uint(f);
  u += 0x7FFFu + ((u >> 16) & 1u);          // round-to-nearest-even
  return (unsigned short)(u >> 16);
}
struct U16x16 { v8u lo, hi; };
static __device__ __forceinline__ v16u join8(v8u lo, v8u hi) {
  U16x16 t{lo, hi};
  return __builtin_bit_cast(v16u, t);
}
static __device__ __forceinline__ v8f wmma_bf16(v16u a, v16u b, v8f c) {
  return __builtin_amdgcn_wmma_f32_16x16x32_bf16(
      false, __builtin_bit_cast(v16bf, a),
      false, __builtin_bit_cast(v16bf, b),
      (short)0, c, false, false);
}

// ---------------------------------------------------------------------------
// f32 -> bf16 cast, 8 elements/thread (n must be a multiple of 8)
// ---------------------------------------------------------------------------
__global__ __launch_bounds__(256) void k_cast_bf16(
    const float* __restrict__ in, unsigned short* __restrict__ out, size_t n8) {
  size_t i = (size_t)blockIdx.x * blockDim.x + threadIdx.x;
  if (i >= n8) return;
  size_t b0 = i * 8;
  float4 a = ((const float4*)(in + b0))[0];
  float4 b = ((const float4*)(in + b0))[1];
  v8u o8;
  o8[0] = f32_to_bf16(a.x); o8[1] = f32_to_bf16(a.y);
  o8[2] = f32_to_bf16(a.z); o8[3] = f32_to_bf16(a.w);
  o8[4] = f32_to_bf16(b.x); o8[5] = f32_to_bf16(b.y);
  o8[6] = f32_to_bf16(b.z); o8[7] = f32_to_bf16(b.w);
  *(v8u*)(out + b0) = o8;
}

// Transpose-cast: out[c*Rpad + r] = bf16(in[r*C + c]); zero-fill r in [R,Rpad)
__global__ __launch_bounds__(256) void k_transpose_cast(
    const float* __restrict__ in, unsigned short* __restrict__ out,
    int R, int C, int Rpad) {
  __shared__ float t[32][33];
  int r0 = blockIdx.y * 32, c0 = blockIdx.x * 32;
  int tx = threadIdx.x & 31, ty = threadIdx.x >> 5;  // 32x8
  for (int i = ty; i < 32; i += 8) {
    int r = r0 + i, c = c0 + tx;
    t[i][tx] = (r < R && c < C) ? in[(size_t)r * C + c] : 0.f;
  }
  __syncthreads();
  for (int i = ty; i < 32; i += 8) {
    int c = c0 + i, r = r0 + tx;
    if (c < C && r < Rpad) out[(size_t)c * Rpad + r] = f32_to_bf16(t[tx][i]);
  }
}

// ---------------------------------------------------------------------------
// LayerNorm (strided input, padded bf16 output with zero tail)
// ---------------------------------------------------------------------------
__global__ __launch_bounds__(256) void k_layernorm(
    const float* __restrict__ in, int in_stride,
    const float* __restrict__ w, const float* __restrict__ b,
    float* __restrict__ outf, unsigned short* __restrict__ outb,
    int ncols, int out_stride) {
  int row = blockIdx.x;
  const float* x = in + (size_t)row * in_stride;
  __shared__ float red0[8], red1[8];
  float s = 0.f, s2 = 0.f;
  for (int c = threadIdx.x; c < ncols; c += blockDim.x) {
    float v = x[c]; s += v; s2 += v * v;
  }
  int lane = threadIdx.x & 31, wv = threadIdx.x >> 5;
  for (int o = 16; o > 0; o >>= 1) {
    s  += __shfl_down(s,  o, 32);
    s2 += __shfl_down(s2, o, 32);
  }
  if (lane == 0) { red0[wv] = s; red1[wv] = s2; }
  __syncthreads();
  if (threadIdx.x == 0) {
    float a = 0.f, a2 = 0.f;
    for (int i = 0; i < 8; ++i) { a += red0[i]; a2 += red1[i]; }
    red0[0] = a; red1[0] = a2;
  }
  __syncthreads();
  float mean = red0[0] / (float)ncols;
  float var  = red1[0] / (float)ncols - mean * mean;
  float rstd = rsqrtf(var + 1e-5f);
  for (int c = threadIdx.x; c < out_stride; c += blockDim.x) {
    float v = 0.f;
    if (c < ncols) v = (x[c] - mean) * rstd * w[c] + b[c];
    if (outf && c < ncols) outf[(size_t)row * ncols + c] = v;
    if (outb) outb[(size_t)row * out_stride + c] = f32_to_bf16(v);
  }
}

// ---------------------------------------------------------------------------
// WMMA GEMM: C[MxN] = A[MxK](bf16) * Bt[NxK](bf16, pre-transposed)
// 256 thr = 8 waves; tile 128x128x32; wave = 32x64 patch (2x4 accumulators).
// All staging + fragment loads are 128-bit. Requires M%128==0, K%32==0, K%8==0.
// ---------------------------------------------------------------------------
#define GM 128
#define GN 128
#define GK 32
#define LDA 40       // u16 per LDS row: 32 + 8 pad (80B, 16B aligned)
#define GEMM_GELU 1

__global__ __launch_bounds__(256) void k_gemm_bf16(
    const unsigned short* __restrict__ A, const unsigned short* __restrict__ Bt,
    float* __restrict__ Cf, unsigned short* __restrict__ Cb,
    const float* __restrict__ bias, const float* __restrict__ resid,
    int M, int N, int K, int flags) {
  __shared__ unsigned short lA[GM * LDA];
  __shared__ unsigned short lB[GN * LDA];

  const int tid  = threadIdx.x;
  const int lane = tid & 31;
  const int w    = tid >> 5;
  const int wm   = w & 3;      // 4 waves along M
  const int wn   = w >> 2;     // 2 waves along N
  const int half = lane >> 4;
  const int l16  = lane & 15;
  const int m0   = blockIdx.y * GM;
  const int n0   = blockIdx.x * GN;
  const bool bfull = (n0 + GN <= N);

  v8f acc[2][4];
  for (int i = 0; i < 2; ++i)
    for (int j = 0; j < 4; ++j)
      for (int e = 0; e < 8; ++e) acc[i][j][e] = 0.f;

  for (int k0 = 0; k0 < K; k0 += GK) {
    // A tile: 128x32, 512 groups of 8 u16 -> b128 loads/stores
    for (int gg = tid; gg < 512; gg += 256) {
      int r = gg >> 2, co = (gg & 3) * 8;
      *(v8u*)(lA + r * LDA + co) =
          *(const v8u*)(A + (size_t)(m0 + r) * K + k0 + co);
    }
    if (bfull) {
      for (int gg = tid; gg < 512; gg += 256) {
        int r = gg >> 2, co = (gg & 3) * 8;
        *(v8u*)(lB + r * LDA + co) =
            *(const v8u*)(Bt + (size_t)(n0 + r) * K + k0 + co);
      }
    } else {                    // ragged N edge (ckv GEMM only)
      for (int id = tid; id < GN * GK; id += 256) {
        int r = id / GK, c = id % GK;
        int gn = n0 + r;
        lB[r * LDA + c] = (gn < N) ? Bt[(size_t)gn * K + k0 + c] : (unsigned short)0;
      }
    }
    if (k0 + GK < K) {          // global_prefetch_b8 of next K-tile
      __builtin_prefetch(A + (size_t)(m0 + (tid >> 1)) * K + k0 + GK, 0, 1);
      if (n0 + (tid >> 1) < N)
        __builtin_prefetch(Bt + (size_t)(n0 + (tid >> 1)) * K + k0 + GK, 0, 1);
    }
    __syncthreads();

    // fragments: A lane = two contiguous 8-u16 runs (ISA 7.12.2); same for Bt
    v16u af[2], bfr[4];
    for (int mt = 0; mt < 2; ++mt) {
      const unsigned short* p = lA + (wm * 32 + mt * 16 + l16) * LDA + half * 8;
      af[mt] = join8(*(const v8u*)p, *(const v8u*)(p + 16));
    }
    for (int nt = 0; nt < 4; ++nt) {
      const unsigned short* p = lB + (wn * 64 + nt * 16 + l16) * LDA + half * 8;
      bfr[nt] = join8(*(const v8u*)p, *(const v8u*)(p + 16));
    }
    for (int mt = 0; mt < 2; ++mt)
      for (int nt = 0; nt < 4; ++nt)
        acc[mt][nt] = wmma_bf16(af[mt], bfr[nt], acc[mt][nt]);
    __syncthreads();
  }

  // epilogue: C layout row = v + 8*half, col = l16
  for (int mt = 0; mt < 2; ++mt)
    for (int nt = 0; nt < 4; ++nt)
      for (int v = 0; v < 8; ++v) {
        int row = m0 + wm * 32 + mt * 16 + v + 8 * half;
        int col = n0 + wn * 64 + nt * 16 + l16;
        if (row < M && col < N) {
          float x = acc[mt][nt][v];
          if (bias)  x += bias[col];
          if (flags & GEMM_GELU) x = 0.5f * x * (1.f + erff(x * 0.70710678118f));
          if (resid) x += resid[(size_t)row * N + col];
          if (Cf) Cf[(size_t)row * N + col] = x;
          if (Cb) Cb[(size_t)row * N + col] = f32_to_bf16(x);
        }
      }
}

// ---------------------------------------------------------------------------
// RoPE pack kernels
// ---------------------------------------------------------------------------
#define TB_S 2048
#define TB_H 16
#define TB_KVP 1365
#define TB_CKV 1429

static __device__ __forceinline__ float rope_val(const float* base, int j, float pos) {
  float fr  = __powf(10000.f, -(float)(2 * (j & 31)) / 128.f);
  float ang = pos * fr;
  float rot = (j < 32) ? -base[j + 32] : base[j - 32];
  return base[j] * __cosf(ang) + rot * __sinf(ang);
}

__global__ __launch_bounds__(128) void k_build_q(
    const float* __restrict__ Q, unsigned short* __restrict__ qo) { // ->[B,H,S,128]
  int idx = blockIdx.x;
  int s = idx % TB_S, h = (idx / TB_S) % TB_H, b = idx / (TB_S * TB_H);
  int d = threadIdx.x;
  const float* src = Q + ((size_t)b * TB_S + s) * (TB_H * 128) + h * 128;
  float val = (d < 64) ? src[d] : rope_val(src + 64, d - 64, (float)s);
  qo[((size_t)(b * TB_H + h) * TB_S + s) * 128 + d] = f32_to_bf16(val);
}

__global__ __launch_bounds__(128) void k_build_kv(
    const float* __restrict__ KV, const float* __restrict__ ckv,
    unsigned short* __restrict__ ko,   // [B,H,S,128]
    unsigned short* __restrict__ vT) { // [B,H,128,S]  (transposed for PV GEMM)
  int idx = blockIdx.x;
  int s = idx % TB_S, h = (idx / TB_S) % TB_H, b = idx / (TB_S * TB_H);
  int d = threadIdx.x;
  const float* kvsrc = KV + ((size_t)b * TB_S + s) * (TB_H * 192) + h * 192;
  float kval;
  if (d < 64) kval = kvsrc[d];
  else {
    const float* kr = ckv + ((size_t)b * TB_S + s) * TB_CKV + TB_KVP;
    kval = rope_val(kr, d - 64, (float)s);
  }
  int bh = b * TB_H + h;
  ko[((size_t)bh * TB_S + s) * 128 + d] = f32_to_bf16(kval);
  vT[((size_t)bh * 128 + d) * TB_S + s] = f32_to_bf16(kvsrc[64 + d]);
}

// ---------------------------------------------------------------------------
// Causal flash attention, head dim 128; 4 waves x 16 query rows.
// K/V tiles staged by TDM (tensor_load_to_lds + s_wait_tensorcnt) when the
// builtin exists, else vectorized copies. Padded LDS strides keep every
// fragment read a ds_load_b128.
// ---------------------------------------------------------------------------
#define FA_BM 64
#define FA_BN 64
#define FA_LDK 136   // 128 + 8 pad (u16)
#define FA_LDV 72    // 64 + 8 pad
#define FA_LDP 72

struct FlashSmem {
  unsigned short K[FA_BN * FA_LDK];   // [key][d]
  unsigned short V[128 * FA_LDV];     // [d][key]
  unsigned short P[4][16 * FA_LDP];   // per-wave [qrow][key]
};

#if USE_TDM
typedef __attribute__((ext_vector_type(4))) unsigned int v4ui;
typedef __attribute__((ext_vector_type(8))) int v8i;
typedef __attribute__((ext_vector_type(4))) int v4i;

// 2D tile load via Tensor Data Mover (D# per ISA cdna5 §8; 2-byte elements,
// LDS row padding via pad_interval/pad_amount).
static __device__ __forceinline__ void tdm_load_2d(
    unsigned lds_byte_off, const void* gaddr,
    unsigned tensor_d0, unsigned tensor_d1,
    unsigned tile_d0, unsigned tile_d1,
    unsigned stride0, unsigned pad_interval_code, unsigned pad_amount_code) {
  unsigned long long ga = (unsigned long long)(uintptr_t)gaddr;
  v4ui g0;
  g0[0] = 1u;                                        // count=1
  g0[1] = lds_byte_off;
  g0[2] = (unsigned)(ga & 0xFFFFFFFFull);
  g0[3] = (unsigned)((ga >> 32) & 0x01FFFFFFull) | (2u << 30);  // type=2
  v8i g1;
  g1[0] = (int)((1u << 16)                           // data_size = 2B
              | (1u << 20)                           // pad_enable
              | (pad_interval_code << 22) | (pad_amount_code << 25));
  g1[1] = (int)((tensor_d0 & 0xFFFFu) << 16);
  g1[2] = (int)((tensor_d0 >> 16) | ((tensor_d1 & 0xFFFFu) << 16));
  g1[3] = (int)((tensor_d1 >> 16) | (tile_d0 << 16));
  g1[4] = (int)tile_d1;                              // tile_dim2 = 0
  g1[5] = (int)stride0;
  g1[6] = 0;
  g1[7] = 0;
  v4i z4 = {0, 0, 0, 0};
#if __clang_major__ >= 23
  v8i z8 = {0, 0, 0, 0, 0, 0, 0, 0};
  __builtin_amdgcn_tensor_load_to_lds(g0, g1, z4, z4, z8, 0);
#else
  __builtin_amdgcn_tensor_load_to_lds(g0, g1, z4, z4, 0);
#endif
}
#endif  // USE_TDM

__global__ __launch_bounds__(128) void k_flash_attn(
    const unsigned short* __restrict__ q,
    const unsigned short* __restrict__ k,
    const unsigned short* __restrict__ vT,
    unsigned short* __restrict__ o, int S, int H) {
  int bh = blockIdx.y, qb = blockIdx.x;
  int b = bh / H, h = bh % H;
  const size_t base = (size_t)bh * S * 128;

  int tid = threadIdx.x, lane = tid & 31, w = tid >> 5;
  int half = lane >> 4, l16 = lane & 15;

  __shared__ FlashSmem sm;

  // Q fragments straight from global: two global_load_b128 per 32-d chunk
  const int qrow0 = qb * FA_BM + w * 16;
  v16u aQ[4];
  for (int kc = 0; kc < 4; ++kc) {
    const unsigned short* p =
        q + base + (size_t)(qrow0 + l16) * 128 + kc * 32 + half * 8;
    aQ[kc] = join8(*(const v8u*)p, *(const v8u*)(p + 16));
  }

  float mrow[8], lrow[8];
  v8f accO[8];
  for (int vv = 0; vv < 8; ++vv) { mrow[vv] = -3.0e38f; lrow[vv] = 0.f; }
  for (int nt = 0; nt < 8; ++nt)
    for (int e = 0; e < 8; ++e) accO[nt][e] = 0.f;

  const float scale = 0.088388347648f;  // 1/sqrt(128)

  for (int kvb = 0; kvb <= qb; ++kvb) {
#if USE_TDM
    if (tid < 32) {             // wave 0 drives both DMA tiles
      tdm_load_2d((unsigned)(uintptr_t)(void*)sm.K,
                  k + base + (size_t)kvb * FA_BN * 128,
                  128u, (unsigned)S, 128u, FA_BN, 128u,
                  /*64 dwords per row*/5u, /*pad 4 dwords*/3u);
      tdm_load_2d((unsigned)(uintptr_t)(void*)sm.V,
                  vT + (size_t)bh * 128 * S + (size_t)kvb * FA_BN,
                  (unsigned)S, 128u, FA_BN, 128u, (unsigned)S,
                  /*32 dwords per row*/4u, /*pad 4 dwords*/3u);
      __builtin_amdgcn_s_wait_tensorcnt(0);
    }
#else
    for (int gg = tid; gg < 1024; gg += 128) {       // K: 64x128
      int key = gg >> 4, doff = (gg & 15) * 8;
      *(v8u*)(sm.K + key * FA_LDK + doff) =
          *(const v8u*)(k + base + (size_t)(kvb * FA_BN + key) * 128 + doff);
    }
    for (int gg = tid; gg < 1024; gg += 128) {       // V^T: 128x64
      int d = gg >> 3, ko = (gg & 7) * 8;
      *(v8u*)(sm.V + d * FA_LDV + ko) =
          *(const v8u*)(vT + ((size_t)bh * 128 + d) * S + kvb * FA_BN + ko);
    }
#endif
    __syncthreads();

    // --- scores: Q (A-frag) x K^T (B-frag from [key][d]) ---
    v8f sc[4];
    for (int nt = 0; nt < 4; ++nt) {
      for (int e = 0; e < 8; ++e) sc[nt][e] = 0.f;
      for (int kc = 0; kc < 4; ++kc) {
        const unsigned short* p =
            sm.K + (nt * 16 + l16) * FA_LDK + kc * 32 + half * 8;
        v16u bK = join8(*(const v8u*)p, *(const v8u*)(p + 16));
        sc[nt] = wmma_bf16(aQ[kc], bK, sc[nt]);
      }
    }

    // --- online softmax on C fragments ---
    for (int vv = 0; vv < 8; ++vv) {
      int qi = qb * FA_BM + w * 16 + vv + 8 * half;
      float mx = -3.0e38f;
      for (int nt = 0; nt < 4; ++nt) {
        int kj = kvb * FA_BN + nt * 16 + l16;
        float sv = sc[nt][vv] * scale;
        sv = (kj <= qi) ? sv : -3.0e38f;
        sc[nt][vv] = sv;
        mx = fmaxf(mx, sv);
      }
      for (int off = 8; off > 0; off >>= 1) mx = fmaxf(mx, __shfl_xor(mx, off, 16));
      float mnew  = fmaxf(mrow[vv], mx);
      float alpha = __expf(mrow[vv] - mnew);
      float rsum  = 0.f;
      for (int nt = 0; nt < 4; ++nt) {
        float p = __expf(sc[nt][vv] - mnew);
        sc[nt][vv] = p;
        rsum += p;
      }
      for (int off = 8; off > 0; off >>= 1) rsum += __shfl_xor(rsum, off, 16);
      lrow[vv] = lrow[vv] * alpha + rsum;
      mrow[vv] = mnew;
      for (int nt = 0; nt < 8; ++nt) accO[nt][vv] *= alpha;
    }

    // --- P: C-layout -> per-wave LDS -> A-layout (b128 reloads) ---
    for (int nt = 0; nt < 4; ++nt)
      for (int vv = 0; vv < 8; ++vv)
        sm.P[w][(vv + 8 * half) * FA_LDP + nt * 16 + l16] =
            f32_to_bf16(sc[nt][vv]);

    for (int kc = 0; kc < 2; ++kc) {
      const unsigned short* pp = sm.P[w] + l16 * FA_LDP + kc * 32 + half * 8;
      v16u aP = join8(*(const v8u*)pp, *(const v8u*)(pp + 16));
      for (int nt = 0; nt < 8; ++nt) {
        const unsigned short* pv =
            sm.V + (nt * 16 + l16) * FA_LDV + kc * 32 + half * 8;
        v16u bV = join8(*(const v8u*)pv, *(const v8u*)(pv + 16));
        accO[nt] = wmma_bf16(aP, bV, accO[nt]);
      }
    }
    __syncthreads();
  }

  for (int nt = 0; nt < 8; ++nt)
    for (int vv = 0; vv < 8; ++vv) {
      int srow = qb * FA_BM + w * 16 + vv + 8 * half;
      int d    = nt * 16 + l16;
      float val = accO[nt][vv] / lrow[vv];
      o[((size_t)b * S + srow) * (H * 128) + h * 128 + d] = f32_to_bf16(val);
    }
}

// ---------------------------------------------------------------------------
// Host orchestration
// ---------------------------------------------------------------------------
extern "C" void kernel_launch(void* const* d_in, const int* in_sizes, int n_in,
                              void* d_out, int out_size, void* d_ws, size_t ws_size,
                              hipStream_t stream) {
  (void)in_sizes; (void)n_in; (void)out_size; (void)ws_size;
  constexpr int B = 2, S = 2048, D = 2048, H = 16, QP = 1024;
  constexpr int KVP = 1365, RD = 64, CKV = KVP + RD;   // 1429
  constexpr int KP  = 1376;                             // KVP padded to %32
  constexpr int KVD = D + H * 64;                       // 3072
  constexpr int FF = 8192, BS = B * S;                  // 4096

  const float* x    = (const float*)d_in[0];
  const float* anw  = (const float*)d_in[1];
  const float* anb  = (const float*)d_in[2];
  const float* Wdq  = (const float*)d_in[3];
  const float* qlw  = (const float*)d_in[4];
  const float* qlb  = (const float*)d_in[5];
  const float* Wuq  = (const float*)d_in[6];
  const float* Wdkv = (const float*)d_in[7];
  const float* klw  = (const float*)d_in[8];
  const float* klb  = (const float*)d_in[9];
  const float* Wukv = (const float*)d_in[10];
  const float* Wo   = (const float*)d_in[11];
  const float* fnw  = (const float*)d_in[12];
  const float* fnb  = (const float*)d_in[13];
  const float* W1   = (const float*)d_in[14];
  const float* b1   = (const float*)d_in[15];
  const float* W2   = (const float*)d_in[16];
  const float* b2   = (const float*)d_in[17];

  float* out1    = (float*)d_out;
  float* out_ckv = out1 + (size_t)BS * D;

  char* ws = (char*)d_ws;
  size_t off = 0;
  auto alloc = [&](size_t bytes) -> void* {
    void* p = ws + off;
    off = (off + bytes + 255) & ~(size_t)255;
    return p;
  };
  typedef unsigned short u16;
  u16* wdqT  = (u16*)alloc((size_t)QP * D * 2);    // [QP][D]
  u16* wuqT  = (u16*)alloc((size_t)D * QP * 2);    // [D][QP]
  u16* wdkvT = (u16*)alloc((size_t)CKV * D * 2);   // [CKV][D]
  u16* wukvT = (u16*)alloc((size_t)KVD * KP * 2);  // [KVD][KP] (K padded)
  u16* w1T   = (u16*)alloc((size_t)FF * D * 2);    // [FF][D]
  u16* w2T   = (u16*)alloc((size_t)D * FF * 2);    // [D][FF]
  u16* wo_b  = (u16*)alloc((size_t)D * D * 2);     // W_o itself = Bt for o@Wo^T
  u16* h_b   = (u16*)alloc((size_t)BS * D * 2);
  float* cq_f = (float*)alloc((size_t)BS * QP * 4);
  u16* cq_b   = (u16*)alloc((size_t)BS * QP * 2);
  float* Q_f  = (float*)alloc((size_t)BS * D * 4);
  float* ckv_f= (float*)alloc((size_t)BS * CKV * 4);
  u16* kvl_b  = (u16*)alloc((size_t)BS * KP * 2);  // padded K
  float* KV_f = (float*)alloc((size_t)BS * KVD * 4);
  u16* q_b    = (u16*)alloc((size_t)B * H * S * 128 * 2);
  u16* k_b    = (u16*)alloc((size_t)B * H * S * 128 * 2);
  u16* vT_b   = (u16*)alloc((size_t)B * H * 128 * S * 2);
  u16* attn_b = (u16*)alloc((size_t)BS * D * 2);
  float* x1_f = (float*)alloc((size_t)BS * D * 4);
  u16* h2_b   = (u16*)alloc((size_t)BS * D * 2);
  u16* ff1_b  = (u16*)alloc((size_t)BS * FF * 2);

  auto tcast = [&](const float* src, u16* dst, int R, int C, int Rpad) {
    k_transpose_cast<<<dim3((C + 31) / 32, (Rpad + 31) / 32), 256, 0, stream>>>(
        src, dst, R, C, Rpad);
  };
  auto gemm = [&](const u16* A, const u16* Bt, float* Cf, u16* Cb,
                  const float* bias, const float* res, int M, int N, int K, int fl) {
    dim3 g((N + GN - 1) / GN, (M + GM - 1) / GM);
    k_gemm_bf16<<<g, 256, 0, stream>>>(A, Bt, Cf, Cb, bias, res, M, N, K, fl);
  };

  // weight prep: Bt = B^T (bf16), K padded where needed
  tcast(Wdq,  wdqT,  D,   QP,  D);
  tcast(Wuq,  wuqT,  QP,  D,   QP);
  tcast(Wdkv, wdkvT, D,   CKV, D);
  tcast(Wukv, wukvT, KVP, KVD, KP);
  tcast(W1,   w1T,   D,   FF,  D);
  tcast(W2,   w2T,   FF,  D,   FF);
  k_cast_bf16<<<dim3((unsigned)(((size_t)D * D / 8 + 255) / 256)), 256, 0, stream>>>(
      Wo, wo_b, (size_t)D * D / 8);

  // h = LN(x)
  k_layernorm<<<BS, 256, 0, stream>>>(x, D, anw, anb, nullptr, h_b, D, D);
  // cq = LN(h @ W_dq)
  gemm(h_b, wdqT, cq_f, nullptr, nullptr, nullptr, BS, QP, D, 0);
  k_layernorm<<<BS, 256, 0, stream>>>(cq_f, QP, qlw, qlb, nullptr, cq_b, QP, QP);
  // Q = cq @ W_uq
  gemm(cq_b, wuqT, Q_f, nullptr, nullptr, nullptr, BS, D, QP, 0);
  // ckv = h @ W_dkv  (output #2)
  gemm(h_b, wdkvT, ckv_f, nullptr, nullptr, nullptr, BS, CKV, D, 0);
  hipMemcpyAsync(out_ckv, ckv_f, (size_t)BS * CKV * 4, hipMemcpyDeviceToDevice, stream);
  // kv_lora = LN(ckv[:, :KVP]) -> padded bf16
  k_layernorm<<<BS, 256, 0, stream>>>(ckv_f, CKV, klw, klb, nullptr, kvl_b, KVP, KP);
  // KV = kv_lora @ W_ukv
  gemm(kvl_b, wukvT, KV_f, nullptr, nullptr, nullptr, BS, KVD, KP, 0);
  // pack q/k/v (V transposed)
  k_build_q <<<B * H * S, 128, 0, stream>>>(Q_f, q_b);
  k_build_kv<<<B * H * S, 128, 0, stream>>>(KV_f, ckv_f, k_b, vT_b);
  // attention
  k_flash_attn<<<dim3(S / FA_BM, B * H), 128, 0, stream>>>(q_b, k_b, vT_b, attn_b, S, H);
  // x1 = x + attn @ W_o^T   (Bt = W_o)
  gemm(attn_b, wo_b, x1_f, nullptr, nullptr, x, BS, D, D, 0);
  // h2 = LN(x1)
  k_layernorm<<<BS, 256, 0, stream>>>(x1_f, D, fnw, fnb, nullptr, h2_b, D, D);
  // ff1 = gelu(h2 @ W1 + b1)
  gemm(h2_b, w1T, nullptr, ff1_b, b1, nullptr, BS, FF, D, GEMM_GELU);
  // out = x1 + ff1 @ W2 + b2
  gemm(ff1_b, w2T, out1, nullptr, b2, x1_f, BS, D, FF, 0);
}
